// ARMA_49297634623854
// MI455X (gfx1250) — matrix operand order repaired and checked by reading, not compiled, not run on hardware
//
#include <hip/hip_runtime.h>
#include <hip/hip_bf16.h>

typedef __attribute__((ext_vector_type(16))) __bf16 v16bf;
typedef __attribute__((ext_vector_type(8)))  float  v8f;

// ---------- helpers ----------
__device__ inline unsigned short f2bf(float f) {
    unsigned int u = __float_as_uint(f);
    u += 0x7FFFu + ((u >> 16) & 1u);     // round-to-nearest-even
    return (unsigned short)(u >> 16);
}

struct U32x8 { uint4 a, b; };
__device__ inline v16bf pack16(uint4 a, uint4 b) {
    union { U32x8 u; v16bf v; } x;
    x.u.a = a; x.u.b = b;
    return x.v;
}

// ---------- elementwise / setup kernels ----------
__global__ void zero_f32_kernel(float* __restrict__ p, int n) {
    int i = blockIdx.x * blockDim.x + threadIdx.x;
    for (; i < n; i += gridDim.x * blockDim.x) p[i] = 0.0f;
}

__global__ void deg_kernel(const int* __restrict__ ei, const float* __restrict__ w,
                           float* __restrict__ deg, int E) {
    int e = blockIdx.x * blockDim.x + threadIdx.x;
    if (e >= E) return;
    int c = ei[E + e];                    // target i
    atomicAdd(&deg[c], w[e]);
}

__global__ void dinv_kernel(float* __restrict__ deg, int N) {
    int i = blockIdx.x * blockDim.x + threadIdx.x;
    if (i >= N) return;
    float d = deg[i];
    deg[i] = (d > 0.0f) ? __frsqrt_rn(fmaxf(d, 1e-12f)) : 0.0f;
}

__global__ void norm_kernel(const int* __restrict__ ei, const float* __restrict__ w,
                            const float* __restrict__ dinv, float* __restrict__ norm, int E) {
    int e = blockIdx.x * blockDim.x + threadIdx.x;
    if (e >= E) return;
    int r = ei[e], c = ei[E + e];
    norm[e] = dinv[r] * w[e] * dinv[c];
}

__global__ void cvt_bf16_kernel(const float* __restrict__ src, unsigned short* __restrict__ dst, int n) {
    int i = blockIdx.x * blockDim.x + threadIdx.x;
    if (i >= n) return;
    dst[i] = f2bf(src[i]);
}

// WcatT[j*128 + k] = (j<128 ? Wi[k,j] : Wr[k,j-128]) in bf16  (column-major concat weight)
__global__ void build_wcat_kernel(const float* __restrict__ Wi, const float* __restrict__ Wr,
                                  unsigned short* __restrict__ WcatT) {
    int t = blockIdx.x * blockDim.x + threadIdx.x;   // 256*128 threads
    if (t >= 256 * 128) return;
    int j = t >> 7, k = t & 127;
    float v = (j < 128) ? Wi[k * 128 + j] : Wr[k * 128 + (j - 128)];
    WcatT[j * 128 + k] = f2bf(v);
}

// ---------- WMMA GEMM: C[N,256] = A[N,128](bf16) @ Wcat[128,256](bf16, stored transposed) ----------
template<bool FULL>
__device__ inline void gemm_body(const unsigned short* __restrict__ Abf,
                                 float* __restrict__ C, const v16bf* bfrag,
                                 int rowBlock, int colTile, int half, int l16, int nrows) {
    #pragma unroll
    for (int rt = 0; rt < 8; ++rt) {
        int row = rowBlock * 128 + rt * 16 + l16;      // A row M = l16 for both halves
        int rc  = FULL ? row : ((row < nrows) ? row : 0);
        const unsigned short* ap = Abf + (size_t)rc * 128 + half * 8;
        v8f acc = {};
        #pragma unroll
        for (int c = 0; c < 4; ++c) {
            // A 16x32 bf16 layout: lane K = c*32 + half*8 .. +8  and  c*32 + 16 + half*8 .. +8
            uint4 q0 = *(const uint4*)(ap + c * 32);
            uint4 q1 = *(const uint4*)(ap + c * 32 + 16);
            v16bf afrag = pack16(q0, q1);
            acc = __builtin_amdgcn_wmma_f32_16x16x32_bf16(
                      false, afrag, false, bfrag[c], (short)0, acc, false, false);
        }
        // C layout: lane col = l16, VGPR r -> row = rt*16 + 8*half + r
        int mBase = rowBlock * 128 + rt * 16 + 8 * half;
        float* cp = C + (size_t)mBase * 256 + colTile * 16 + l16;
        if (FULL) {
            #pragma unroll
            for (int r = 0; r < 8; ++r) cp[(size_t)r * 256] = acc[r];   // imm-offset stores
        } else {
            #pragma unroll
            for (int r = 0; r < 8; ++r) {
                if (mBase + r < nrows) cp[(size_t)r * 256] = acc[r];
            }
        }
    }
}

__global__ __launch_bounds__(256)
void gemm_wmma_kernel(const unsigned short* __restrict__ Abf,   // [nrows,128] bf16 row-major
                      const unsigned short* __restrict__ BT,    // [256,128]  bf16 (col-major of W)
                      float* __restrict__ C,                    // [nrows,256] f32
                      int nrows, int rowBlocks) {
    const int wave = threadIdx.x >> 5;
    const int lane = threadIdx.x & 31;
    const int gw   = blockIdx.x * 8 + wave;
    const int colTile  = gw & 15;          // 16 column tiles of 16
    const int rowBlock = gw >> 4;          // 128 rows per block
    if (rowBlock >= rowBlocks) return;

    const int half = lane >> 4;            // 0 / 1
    const int l16  = lane & 15;

    // B fragments (32x16 per K-chunk): lane holds col n = colTile*16+l16,
    // K = chunk*32 + half*8 .. +8  and  chunk*32 + 16 + half*8 .. +8 in BT.
    v16bf bfrag[4];
    const unsigned short* bp = BT + (colTile * 16 + l16) * 128 + half * 8;
    #pragma unroll
    for (int c = 0; c < 4; ++c) {
        uint4 q0 = *(const uint4*)(bp + c * 32);
        uint4 q1 = *(const uint4*)(bp + c * 32 + 16);
        bfrag[c] = pack16(q0, q1);
    }

    const bool full = (rowBlock * 128 + 128) <= nrows;   // wave-uniform
    if (full) gemm_body<true >(Abf, C, bfrag, rowBlock, colTile, half, l16, nrows);
    else      gemm_body<false>(Abf, C, bfrag, rowBlock, colTile, half, l16, nrows);
}

// ---------- edge scatter: agg[col] += norm[e] * m[row]  (m = C[:, 0:128]) ----------
__global__ void scatter_kernel(const float* __restrict__ mr, const float* __restrict__ norm,
                               const int* __restrict__ ei, float* __restrict__ agg, int E) {
    int t = blockIdx.x * blockDim.x + threadIdx.x;   // E*32 threads, 4 floats each
    int e = t >> 5, q = t & 31;
    if (e >= E) return;
    int r = ei[e], c = ei[E + e];
    float nv = norm[e];
    float4 m4 = *(const float4*)(mr + (size_t)r * 256 + q * 4);
    float* dst = agg + (size_t)c * 128 + q * 4;
    atomicAdd(dst + 0, nv * m4.x);
    atomicAdd(dst + 1, nv * m4.y);
    atomicAdd(dst + 2, nv * m4.z);
    atomicAdd(dst + 3, nv * m4.w);
}

// ---------- combine1: h = relu(agg + r + b) -> bf16 for next GEMM ----------
__global__ void combine1_kernel(const float* __restrict__ agg, const float* __restrict__ mr,
                                const float* __restrict__ b, unsigned short* __restrict__ hbf, int N) {
    int t = blockIdx.x * blockDim.x + threadIdx.x;   // N*32 threads, 4 floats each
    int n = t >> 5, q = t & 31;
    if (n >= N) return;
    int f = q * 4;
    float4 a4 = *(const float4*)(agg + (size_t)n * 128 + f);
    float4 r4 = *(const float4*)(mr + (size_t)n * 256 + 128 + f);
    float4 b4 = *(const float4*)(b + f);
    float h0 = fmaxf(a4.x + r4.x + b4.x, 0.0f);
    float h1 = fmaxf(a4.y + r4.y + b4.y, 0.0f);
    float h2 = fmaxf(a4.z + r4.z + b4.z, 0.0f);
    float h3 = fmaxf(a4.w + r4.w + b4.w, 0.0f);
    unsigned int lo = (unsigned int)f2bf(h0) | ((unsigned int)f2bf(h1) << 16);
    unsigned int hi = (unsigned int)f2bf(h2) | ((unsigned int)f2bf(h3) << 16);
    uint2 pk; pk.x = lo; pk.y = hi;
    *(uint2*)(hbf + (size_t)n * 128 + f) = pk;
}

// ---------- combine2: node_x = relu(agg + r + b); atomic mean-pool accumulate ----------
__global__ void combine2_kernel(const float* __restrict__ agg, const float* __restrict__ mr,
                                const float* __restrict__ b, const int* __restrict__ batch,
                                float* __restrict__ gsum, int N) {
    int t = blockIdx.x * blockDim.x + threadIdx.x;
    int n = t >> 5, q = t & 31;
    if (n >= N) return;
    int f = q * 4;
    int g = batch[n];
    float4 a4 = *(const float4*)(agg + (size_t)n * 128 + f);
    float4 r4 = *(const float4*)(mr + (size_t)n * 256 + 128 + f);
    float4 b4 = *(const float4*)(b + f);
    float* dst = gsum + (size_t)g * 128 + f;
    atomicAdd(dst + 0, fmaxf(a4.x + r4.x + b4.x, 0.0f));
    atomicAdd(dst + 1, fmaxf(a4.y + r4.y + b4.y, 0.0f));
    atomicAdd(dst + 2, fmaxf(a4.z + r4.z + b4.z, 0.0f));
    atomicAdd(dst + 3, fmaxf(a4.w + r4.w + b4.w, 0.0f));
}

__global__ void count_kernel(const int* __restrict__ batch, float* __restrict__ cnt, int N) {
    int i = blockIdx.x * blockDim.x + threadIdx.x;
    if (i >= N) return;
    atomicAdd(&cnt[batch[i]], 1.0f);
}

// ---------- head: out = relu(mean @ mW1 + mb1) @ mW2 + mb2 ----------
__global__ __launch_bounds__(256)
void head_kernel(const float* __restrict__ gsum, const float* __restrict__ cnt,
                 const float* __restrict__ mW1, const float* __restrict__ mb1,
                 const float* __restrict__ mW2, const float* __restrict__ mb2,
                 float* __restrict__ out) {
    __shared__ float gx[128];
    __shared__ float p0[256];
    __shared__ float p1[256];
    int g = blockIdx.x, t = threadIdx.x;
    float c = fmaxf(cnt[g], 1.0f);
    if (t < 128) gx[t] = gsum[(size_t)g * 128 + t] / c;
    __syncthreads();
    float acc = mb1[t];
    #pragma unroll 4
    for (int k = 0; k < 128; ++k) acc += gx[k] * mW1[k * 256 + t];
    float h = fmaxf(acc, 0.0f);
    p0[t] = h * mW2[t * 2 + 0];
    p1[t] = h * mW2[t * 2 + 1];
    __syncthreads();
    for (int s = 128; s > 0; s >>= 1) {
        if (t < s) { p0[t] += p0[t + s]; p1[t] += p1[t + s]; }
        __syncthreads();
    }
    if (t == 0) {
        out[g * 2 + 0] = p0[0] + mb2[0];
        out[g * 2 + 1] = p1[0] + mb2[1];
    }
}

// ---------- launch ----------
extern "C" void kernel_launch(void* const* d_in, const int* in_sizes, int n_in,
                              void* d_out, int out_size, void* d_ws, size_t ws_size,
                              hipStream_t stream) {
    const float* x   = (const float*)d_in[0];
    const int*   ei  = (const int*)d_in[1];
    const float* ea  = (const float*)d_in[2];
    const int*   bat = (const int*)d_in[3];
    const float* Wi1 = (const float*)d_in[4];
    const float* Wr1 = (const float*)d_in[5];
    const float* b1  = (const float*)d_in[6];
    const float* Wi2 = (const float*)d_in[7];
    const float* Wr2 = (const float*)d_in[8];
    const float* b2  = (const float*)d_in[9];
    const float* mW1 = (const float*)d_in[10];
    const float* mb1 = (const float*)d_in[11];
    const float* mW2 = (const float*)d_in[12];
    const float* mb2 = (const float*)d_in[13];
    float* out = (float*)d_out;

    const int N = in_sizes[0] / 128;
    const int E = in_sizes[2];
    const int G = out_size / 2;

    // workspace carve-up (256B aligned)
    char* ws = (char*)d_ws;
    size_t off = 0;
    auto carve = [&](size_t bytes) { char* p = ws + off; off = (off + bytes + 255) & ~(size_t)255; return p; };
    float*          deg   = (float*)carve((size_t)N * 4);            // -> dinv in place
    float*          norm  = (float*)carve((size_t)E * 4);
    unsigned short* xbf   = (unsigned short*)carve((size_t)N * 128 * 2);
    unsigned short* h1bf  = (unsigned short*)carve((size_t)N * 128 * 2);
    unsigned short* Wc1   = (unsigned short*)carve((size_t)256 * 128 * 2);
    unsigned short* Wc2   = (unsigned short*)carve((size_t)256 * 128 * 2);
    float*          mr    = (float*)carve((size_t)N * 256 * 4);      // [m | r], reused both layers
    float*          agg   = (float*)carve((size_t)N * 128 * 4);      // reused both layers
    float*          gsum  = (float*)carve((size_t)G * 128 * 4);
    float*          cnt   = (float*)carve((size_t)G * 4);
    (void)ws_size; (void)n_in;

    const int B = 256;
    auto cdiv = [](int a, int b) { return (a + b - 1) / b; };
    const int rowBlocks  = cdiv(N, 128);
    const int gemmBlocks = cdiv(rowBlocks * 16, 8);

    // normalization
    zero_f32_kernel<<<1024, B, 0, stream>>>(deg, N);
    zero_f32_kernel<<<256,  B, 0, stream>>>(gsum, G * 128);
    zero_f32_kernel<<<8,    B, 0, stream>>>(cnt, G);
    deg_kernel <<<cdiv(E, B), B, 0, stream>>>(ei, ea, deg, E);
    dinv_kernel<<<cdiv(N, B), B, 0, stream>>>(deg, N);
    norm_kernel<<<cdiv(E, B), B, 0, stream>>>(ei, ea, deg, norm, E);

    // precision staging
    cvt_bf16_kernel <<<cdiv(N * 128, B), B, 0, stream>>>(x, xbf, N * 128);
    build_wcat_kernel<<<cdiv(256 * 128, B), B, 0, stream>>>(Wi1, Wr1, Wc1);
    build_wcat_kernel<<<cdiv(256 * 128, B), B, 0, stream>>>(Wi2, Wr2, Wc2);

    // layer 1
    gemm_wmma_kernel<<<gemmBlocks, B, 0, stream>>>(xbf, Wc1, mr, N, rowBlocks);
    zero_f32_kernel<<<4096, B, 0, stream>>>(agg, N * 128);
    scatter_kernel <<<cdiv(E * 32, B), B, 0, stream>>>(mr, norm, ei, agg, E);
    combine1_kernel<<<cdiv(N * 32, B), B, 0, stream>>>(agg, mr, b1, h1bf, N);

    // layer 2
    gemm_wmma_kernel<<<gemmBlocks, B, 0, stream>>>(h1bf, Wc2, mr, N, rowBlocks);
    zero_f32_kernel<<<4096, B, 0, stream>>>(agg, N * 128);
    scatter_kernel <<<cdiv(E * 32, B), B, 0, stream>>>(mr, norm, ei, agg, E);
    combine2_kernel<<<cdiv(N * 32, B), B, 0, stream>>>(agg, mr, b2, bat, gsum, N);
    count_kernel   <<<cdiv(N, B), B, 0, stream>>>(bat, cnt, N);

    // pooled head
    head_kernel<<<G, B, 0, stream>>>(gsum, cnt, mW1, mb1, mW2, mb2, out);
}
